// Stage_68590627717158
// MI455X (gfx1250) — matrix operand order repaired and checked by reading, not compiled, not run on hardware
//
#include <hip/hip_runtime.h>
#include <math.h>

#define NPTS 65536
#define KNNK 16
#define DCH  96
#define HCH  384
#define HDCH 256

typedef __attribute__((ext_vector_type(16))) __bf16 v16bf;
typedef __attribute__((ext_vector_type(8)))  float  v8f;

struct B32x2 { uint4 a, b; };          // 32 bytes == one 16x(bf16) fragment

static __device__ __forceinline__ v16bf frag_from2(uint4 a, uint4 b) {
    B32x2 t{a, b};
    return __builtin_bit_cast(v16bf, t);
}

// B fragment (32x16) from pre-packed bf16 weights: fragment-contiguous layout
static __device__ __forceinline__ v16bf loadB_p(const __bf16* Wp, int Nt, int kc,
                                                int nt, int lane) {
    const uint4* p = (const uint4*)Wp + (size_t)((kc * Nt + nt) * 32 + lane) * 2;
    return frag_from2(p[0], p[1]);
}

// A fragment (16x32) from LDS bf16 row-major tile: two ds_load_b128 per lane
// (row-major 16-bit tile -> fragment via plain B128 loads, ISA §10.9)
static __device__ __forceinline__ v16bf loadA_s(const __bf16* T, int lda, int kbase,
                                                int lane) {
    int M = lane & 15, g = lane >> 4;
    const __bf16* base = T + M * lda + kbase + 8 * g;
    return frag_from2(*(const uint4*)base, *(const uint4*)(base + 16));
}

static __device__ __forceinline__ unsigned pk2(float a, float b) {
    unsigned short lo = __builtin_bit_cast(unsigned short, (__bf16)a);
    unsigned short hi = __builtin_bit_cast(unsigned short, (__bf16)b);
    return (unsigned)lo | ((unsigned)hi << 16);
}
static __device__ __forceinline__ uint2 pk4(float4 v) {
    return make_uint2(pk2(v.x, v.y), pk2(v.z, v.w));
}
static __device__ __forceinline__ float gelu_exact(float x) {
    return 0.5f * x * (1.0f + erff(x * 0.70710678118654752f));
}

// ---------- weight pre-pack: W[K x Nn] fp32 -> bf16 fragment order ----------
// out[((kc*Nt+nt)*32 + lane)*16 + h] = bf16( W[(kc*32 + (h&7) + 8*(lane>>4) + 16*(h>>3)) * Nn
//                                            + nt*16 + (lane&15)] )
__global__ __launch_bounds__(256) void pack_kernel(const float* __restrict__ W,
                                                   __bf16* __restrict__ out,
                                                   int K, int Nn) {
    int tid = blockIdx.x * blockDim.x + threadIdx.x;
    if (tid >= K * Nn) return;
    int h    = tid & 15;
    int lane = (tid >> 4) & 31;
    int rest = tid >> 9;
    int Nt   = Nn >> 4;
    int nt   = rest % Nt;
    int kc   = rest / Nt;
    int k = kc * 32 + (h & 7) + 8 * (lane >> 4) + 16 * (h >> 3);
    int n = nt * 16 + (lane & 15);
    out[tid] = (__bf16)W[k * Nn + n];
}

// ---------- neighbor embedding ----------
// One wave = 2 points. Each lane runs the 7->16->32 MLP for one neighbor-row
// (VALU), rows land in LDS bf16 tiles; 32->96 layer + K-maxpool done with WMMA.
__global__ __launch_bounds__(32) void ne_kernel(
    const float* __restrict__ x, const float* __restrict__ xyz, const int* __restrict__ knn,
    const float* __restrict__ w1, const float* __restrict__ g1, const float* __restrict__ b1,
    const float* __restrict__ w2, const float* __restrict__ g2, const float* __restrict__ b2,
    const __bf16* __restrict__ w3p, const float* __restrict__ ng, const float* __restrict__ nb,
    float* __restrict__ f) {
    __shared__ float sw1[7 * 16], sg1[16], sb1[16];
    __shared__ float sw2[16 * 32], sg2[32], sb2[32];
    __shared__ __align__(16) __bf16 A2[2][16 * 32];

    int lane = threadIdx.x;
    for (int i = lane; i < 7 * 16;  i += 32) sw1[i] = w1[i];
    for (int i = lane; i < 16;      i += 32) { sg1[i] = g1[i]; sb1[i] = b1[i]; }
    for (int i = lane; i < 16 * 32; i += 32) sw2[i] = w2[i];
    sg2[lane] = g2[lane]; sb2[lane] = b2[lane];
    __syncthreads();

    int half = lane >> 4;                    // which of the 2 points
    int k    = lane & 15;                    // neighbor index
    int p    = blockIdx.x * 2 + half;
    int j    = knn[p * KNNK + k];

    float in7[7];
    in7[0] = xyz[j * 3 + 0] - xyz[p * 3 + 0];
    in7[1] = xyz[j * 3 + 1] - xyz[p * 3 + 1];
    in7[2] = xyz[j * 3 + 2] - xyz[p * 3 + 2];
    in7[3] = x[j * 4 + 0]; in7[4] = x[j * 4 + 1];
    in7[5] = x[j * 4 + 2]; in7[6] = x[j * 4 + 3];

    float h1[16];
    #pragma unroll
    for (int c = 0; c < 16; ++c) {
        float s = 0.f;
        #pragma unroll
        for (int i = 0; i < 7; ++i) s += in7[i] * sw1[i * 16 + c];
        h1[c] = gelu_exact(s * sg1[c] + sb1[c]);       // BN then GELU (ref order)
    }
    float h2[32];
    #pragma unroll
    for (int c = 0; c < 32; ++c) {
        float s = 0.f;
        #pragma unroll
        for (int i = 0; i < 16; ++i) s += h1[i] * sw2[i * 32 + c];
        h2[c] = gelu_exact(s * sg2[c] + sb2[c]);
    }
    // write this neighbor-row (32 bf16 = 4x b128) into the point's LDS A tile
    uint4* row = (uint4*)&A2[half][k * 32];
    #pragma unroll
    for (int q = 0; q < 4; ++q) {
        uint4 u;
        u.x = pk2(h2[q * 8 + 0], h2[q * 8 + 1]);
        u.y = pk2(h2[q * 8 + 2], h2[q * 8 + 3]);
        u.z = pk2(h2[q * 8 + 4], h2[q * 8 + 5]);
        u.w = pk2(h2[q * 8 + 6], h2[q * 8 + 7]);
        row[q] = u;
    }
    __syncthreads();

    int nl = lane & 15, g = lane >> 4;
    #pragma unroll
    for (int pp_ = 0; pp_ < 2; ++pp_) {
        v16bf a = loadA_s(A2[pp_], 32, 0, lane);
        int pt = blockIdx.x * 2 + pp_;
        for (int nt = 0; nt < DCH / 16; ++nt) {
            v8f acc = {};
            v16bf b = loadB_p(w3p, DCH / 16, 0, nt, lane);
            acc = __builtin_amdgcn_wmma_f32_16x16x32_bf16(false, a, false, b,
                                                          (short)0, acc, false, false);
            // max over the 16 neighbor-rows: 8 in-lane + partner lane (g^1)
            float m = acc[0];
            #pragma unroll
            for (int r = 1; r < 8; ++r) m = fmaxf(m, acc[r]);
            m = fmaxf(m, __shfl_xor(m, 16, 32));
            if (lane < 16) {
                int n = nt * 16 + nl;
                f[pt * DCH + n] = m * ng[n] + nb[n];
            }
        }
    }
    (void)g;
}

// ---------- fused MLP: f += BN( gelu(f@w1 + b1) @ w2 ), one wave per 16 rows ----------
__global__ __launch_bounds__(32) void mlp_kernel(
    float* __restrict__ f, const __bf16* __restrict__ w1p, const float* __restrict__ b1,
    const __bf16* __restrict__ w2p, const float* __restrict__ gg, const float* __restrict__ bb) {
    __shared__ __align__(16) __bf16 Af[16 * DCH];
    __shared__ __align__(16) __bf16 T[16 * HCH];
    int lane = threadIdx.x;
    int row0 = blockIdx.x * 16;
    int g = lane >> 4, nl = lane & 15;

    // stage f tile (rows contiguous) -> bf16 LDS, b128 loads + packed cvt
    const float4* src = (const float4*)(f + (size_t)row0 * DCH);
    uint2* dst = (uint2*)Af;
    #pragma unroll
    for (int i = 0; i < (16 * DCH / 4) / 32; ++i)        // 12 iters
        dst[i * 32 + lane] = pk4(src[i * 32 + lane]);
    __syncthreads();

    v16bf a1[3];
    #pragma unroll
    for (int kc = 0; kc < 3; ++kc) a1[kc] = loadA_s(Af, DCH, kc * 32, lane);

    // GEMM1: [16x96]@[96x384] + bias + exact GELU -> bf16 LDS
    for (int nt = 0; nt < HCH / 16; ++nt) {
        v8f acc = {};
        #pragma unroll
        for (int kc = 0; kc < 3; ++kc) {
            v16bf bf = loadB_p(w1p, HCH / 16, kc, nt, lane);
            acc = __builtin_amdgcn_wmma_f32_16x16x32_bf16(false, a1[kc], false, bf,
                                                          (short)0, acc, false, false);
        }
        int n = nt * 16 + nl;
        float bias = b1[n];
        #pragma unroll
        for (int r = 0; r < 8; ++r)
            T[(r + 8 * g) * HCH + n] = (__bf16)gelu_exact(acc[r] + bias);
    }
    __syncthreads();

    v16bf a2[HCH / 32];
    #pragma unroll
    for (int kc = 0; kc < HCH / 32; ++kc) a2[kc] = loadA_s(T, HCH, kc * 32, lane);

    // GEMM2: [16x384]@[384x96], BN + residual into f
    for (int nt = 0; nt < DCH / 16; ++nt) {
        v8f acc = {};
        #pragma unroll
        for (int kc = 0; kc < HCH / 32; ++kc) {
            v16bf bf = loadB_p(w2p, DCH / 16, kc, nt, lane);
            acc = __builtin_amdgcn_wmma_f32_16x16x32_bf16(false, a2[kc], false, bf,
                                                          (short)0, acc, false, false);
        }
        int n = nt * 16 + nl;
        float gn = gg[n], bn_ = bb[n];
        #pragma unroll
        for (int r = 0; r < 8; ++r) {
            int row = row0 + r + 8 * g;
            f[(size_t)row * DCH + n] += acc[r] * gn + bn_;
        }
    }
}

// ---------- LFP projection: y = f @ W (96x96), one wave per 16 rows ----------
__global__ __launch_bounds__(32) void lfp_proj_kernel(
    const float* __restrict__ f, const __bf16* __restrict__ wp, float* __restrict__ y) {
    __shared__ __align__(16) __bf16 Af[16 * DCH];
    int lane = threadIdx.x;
    int row0 = blockIdx.x * 16;
    int g = lane >> 4, nl = lane & 15;

    const float4* src = (const float4*)(f + (size_t)row0 * DCH);
    uint2* dst = (uint2*)Af;
    #pragma unroll
    for (int i = 0; i < (16 * DCH / 4) / 32; ++i)
        dst[i * 32 + lane] = pk4(src[i * 32 + lane]);
    __syncthreads();

    v16bf a1[3];
    #pragma unroll
    for (int kc = 0; kc < 3; ++kc) a1[kc] = loadA_s(Af, DCH, kc * 32, lane);

    for (int nt = 0; nt < DCH / 16; ++nt) {
        v8f acc = {};
        #pragma unroll
        for (int kc = 0; kc < 3; ++kc) {
            v16bf bf = loadB_p(wp, DCH / 16, kc, nt, lane);
            acc = __builtin_amdgcn_wmma_f32_16x16x32_bf16(false, a1[kc], false, bf,
                                                          (short)0, acc, false, false);
        }
        int n = nt * 16 + nl;
        #pragma unroll
        for (int r = 0; r < 8; ++r)
            y[(size_t)(row0 + r + 8 * g) * DCH + n] = acc[r];
    }
}

// ---------- LFP edge: f += BN( max_k y[knn[n,k]] - y[n] ), thread per (n,d) ----------
__global__ __launch_bounds__(256) void lfp_edge_kernel(
    float* __restrict__ f, const float* __restrict__ y, const int* __restrict__ knn,
    const float* __restrict__ gg, const float* __restrict__ bb) {
    int idx = blockIdx.x * blockDim.x + threadIdx.x;
    if (idx >= NPTS * DCH) return;
    int n = idx / DCH, d = idx - n * DCH;
    float m = -3.402823466e38f;
    #pragma unroll
    for (int k = 0; k < KNNK; ++k) {
        int j = knn[n * KNNK + k];
        m = fmaxf(m, y[(size_t)j * DCH + d]);      // coalesced along d
    }
    f[idx] += (m - y[idx]) * gg[d] + bb[d];
}

// ---------- head: out = BN(f) @ pp_w (96x256), one wave per 16 rows ----------
__global__ __launch_bounds__(32) void final_kernel(
    const float* __restrict__ f, const float* __restrict__ pg, const float* __restrict__ pb,
    const __bf16* __restrict__ pwp, float* __restrict__ out) {
    __shared__ __align__(16) __bf16 Af[16 * DCH];
    int lane = threadIdx.x;
    int row0 = blockIdx.x * 16;
    int g = lane >> 4, nl = lane & 15;

    // stage with BN folded in (float4 spans 4 consecutive channels, 96%4==0)
    const float4* src = (const float4*)(f + (size_t)row0 * DCH);
    uint2* dst = (uint2*)Af;
    #pragma unroll
    for (int i = 0; i < (16 * DCH / 4) / 32; ++i) {
        int flat = (i * 32 + lane) * 4;
        int col  = flat % DCH;
        float4 v = src[i * 32 + lane];
        v.x = v.x * pg[col + 0] + pb[col + 0];
        v.y = v.y * pg[col + 1] + pb[col + 1];
        v.z = v.z * pg[col + 2] + pb[col + 2];
        v.w = v.w * pg[col + 3] + pb[col + 3];
        dst[i * 32 + lane] = pk4(v);
    }
    __syncthreads();

    v16bf a1[3];
    #pragma unroll
    for (int kc = 0; kc < 3; ++kc) a1[kc] = loadA_s(Af, DCH, kc * 32, lane);

    for (int nt = 0; nt < HDCH / 16; ++nt) {
        v8f acc = {};
        #pragma unroll
        for (int kc = 0; kc < 3; ++kc) {
            v16bf bf = loadB_p(pwp, HDCH / 16, kc, nt, lane);
            acc = __builtin_amdgcn_wmma_f32_16x16x32_bf16(false, a1[kc], false, bf,
                                                          (short)0, acc, false, false);
        }
        int n = nt * 16 + nl;
        #pragma unroll
        for (int r = 0; r < 8; ++r)
            out[(size_t)(row0 + r + 8 * g) * HDCH + n] = acc[r];
    }
}

// ---------- launch ----------
extern "C" void kernel_launch(void* const* d_in, const int* in_sizes, int n_in,
                              void* d_out, int out_size, void* d_ws, size_t ws_size,
                              hipStream_t stream) {
    const float* x     = (const float*)d_in[0];
    const float* xyz   = (const float*)d_in[1];
    const int*   knn   = (const int*)  d_in[2];
    const float* ne_w1 = (const float*)d_in[3];
    const float* ne_g1 = (const float*)d_in[4];
    const float* ne_b1 = (const float*)d_in[5];
    const float* ne_w2 = (const float*)d_in[6];
    const float* ne_g2 = (const float*)d_in[7];
    const float* ne_b2 = (const float*)d_in[8];
    const float* ne_w3 = (const float*)d_in[9];
    const float* nbr_g = (const float*)d_in[10];
    const float* nbr_b = (const float*)d_in[11];
    const float* m0_w1 = (const float*)d_in[12];
    const float* m0_b1 = (const float*)d_in[13];
    const float* m0_w2 = (const float*)d_in[14];
    const float* m0_g  = (const float*)d_in[15];
    const float* m0_b  = (const float*)d_in[16];
    const float* lfp_w = (const float*)d_in[17];
    const float* lfp_g = (const float*)d_in[18];
    const float* lfp_b = (const float*)d_in[19];
    const float* ms_w1 = (const float*)d_in[20];
    const float* ms_b1 = (const float*)d_in[21];
    const float* ms_w2 = (const float*)d_in[22];
    const float* ms_g  = (const float*)d_in[23];
    const float* ms_b  = (const float*)d_in[24];
    const float* pp_g  = (const float*)d_in[25];
    const float* pp_b  = (const float*)d_in[26];
    const float* pp_w  = (const float*)d_in[27];

    float* f = (float*)d_ws;                                   // [N,96]
    float* y = f + (size_t)NPTS * DCH;                         // [N,96]
    __bf16* pk = (__bf16*)(y + (size_t)NPTS * DCH);            // packed weights

    __bf16* p_m0w1 = pk;                 // 96*384
    __bf16* p_m0w2 = p_m0w1 + 36864;     // 384*96
    __bf16* p_msw1 = p_m0w2 + 36864;     // 2 x 96*384
    __bf16* p_msw2 = p_msw1 + 2 * 36864; // 2 x 384*96
    __bf16* p_lfp  = p_msw2 + 2 * 36864; // 4 x 96*96
    __bf16* p_pp   = p_lfp + 4 * 9216;   // 96*256
    __bf16* p_new3 = p_pp + 24576;       // 32*96

    // pre-pack all GEMM weights into bf16 fragment order (tiny, deterministic)
    pack_kernel<<<(36864 + 255) / 256, 256, 0, stream>>>(m0_w1, p_m0w1, DCH, HCH);
    pack_kernel<<<(36864 + 255) / 256, 256, 0, stream>>>(m0_w2, p_m0w2, HCH, DCH);
    for (int j = 0; j < 2; ++j) {
        pack_kernel<<<(36864 + 255) / 256, 256, 0, stream>>>(
            ms_w1 + (size_t)j * DCH * HCH, p_msw1 + (size_t)j * 36864, DCH, HCH);
        pack_kernel<<<(36864 + 255) / 256, 256, 0, stream>>>(
            ms_w2 + (size_t)j * HCH * DCH, p_msw2 + (size_t)j * 36864, HCH, DCH);
    }
    for (int i = 0; i < 4; ++i)
        pack_kernel<<<(9216 + 255) / 256, 256, 0, stream>>>(
            lfp_w + (size_t)i * DCH * DCH, p_lfp + (size_t)i * 9216, DCH, DCH);
    pack_kernel<<<(24576 + 255) / 256, 256, 0, stream>>>(pp_w, p_pp, DCH, HDCH);
    pack_kernel<<<(3072 + 255) / 256, 256, 0, stream>>>(ne_w3, p_new3, 32, DCH);

    ne_kernel<<<NPTS / 2, 32, 0, stream>>>(x, xyz, knn,
        ne_w1, ne_g1, ne_b1, ne_w2, ne_g2, ne_b2, p_new3, nbr_g, nbr_b, f);

    mlp_kernel<<<NPTS / 16, 32, 0, stream>>>(f, p_m0w1, m0_b1, p_m0w2, m0_g, m0_b);

    for (int i = 0; i < 4; ++i) {
        lfp_proj_kernel<<<NPTS / 16, 32, 0, stream>>>(f, p_lfp + (size_t)i * 9216, y);
        lfp_edge_kernel<<<(NPTS * DCH + 255) / 256, 256, 0, stream>>>(
            f, y, knn, lfp_g + i * DCH, lfp_b + i * DCH);
        if (i & 1) {
            int j = i / 2;
            mlp_kernel<<<NPTS / 16, 32, 0, stream>>>(
                f, p_msw1 + (size_t)j * 36864, ms_b1 + j * HCH,
                p_msw2 + (size_t)j * 36864, ms_g + j * DCH, ms_b + j * DCH);
        }
    }

    final_kernel<<<NPTS / 16, 32, 0, stream>>>(f, pp_g, pp_b, p_pp, (float*)d_out);
}
